// GGNN_39384850104697
// MI455X (gfx1250) — matrix operand order repaired and checked by reading, not compiled, not run on hardware
//
#include <hip/hip_runtime.h>
#include <hip/hip_bf16.h>
#include <stdint.h>

typedef __attribute__((ext_vector_type(16))) _Float16 v16h;
typedef __attribute__((ext_vector_type(8)))  _Float16 v8h;
typedef __attribute__((ext_vector_type(8)))  float    v8f;

#define KDIM 256   // inner dim of every GEMM (IN == OUT == 256)

// ---------------------------------------------------------------------------
// WMMA GEMM:  C[M x Ncols] = A[M x 256](f16) @ W[Ncols x 256](f16)^T + bias
// One wave computes a 32(M) x 64(N) tile: 2 A-frags, 4 B-frags shared by both
// M-subtiles -> 8 WMMAs per 12 b128 loads per k-step (K-loop of 8x32).
// A/B lanes use contiguous K (k = 16*(lane/16) + s); f32 C 16x16 per CDNA5 ISA 7.12.2.
// ---------------------------------------------------------------------------
__global__ __launch_bounds__(256)
void ggnn_wmma_gemm(const _Float16* __restrict__ A,
                    const _Float16* __restrict__ W,
                    const float* __restrict__ bias,
                    float* __restrict__ C,
                    int M, int Ncols) {
  const int lane = threadIdx.x & 31;
  const int wave = threadIdx.x >> 5;
  const int mt   = blockIdx.x * 8 + wave;          // 32-row tile index (wave-uniform)
  if (mt * 32 >= M) return;                        // uniform per wave -> EXEC all-1s
  const int n0   = blockIdx.y * 64;

  const int mlane = lane & 15;
  const int kbA   = (lane < 16) ? 0 : 16;          // A-frag K base (contiguous K, 16 per half-wave)
  const int kbB   = (lane < 16) ? 0 : 16;          // B-frag K base (contiguous K, 16 per half-wave)

  const _Float16* __restrict__ arow0 = A + (size_t)(mt * 32 + mlane) * KDIM;
  const _Float16* __restrict__ arow1 = arow0 + (size_t)16 * KDIM;
  const _Float16* __restrict__ w0 = W + (size_t)(n0 +  0 + mlane) * KDIM + kbB;
  const _Float16* __restrict__ w1 = W + (size_t)(n0 + 16 + mlane) * KDIM + kbB;
  const _Float16* __restrict__ w2 = W + (size_t)(n0 + 32 + mlane) * KDIM + kbB;
  const _Float16* __restrict__ w3 = W + (size_t)(n0 + 48 + mlane) * KDIM + kbB;

  v8f a00 = {}, a01 = {}, a02 = {}, a03 = {};      // M-subtile 0, N-subtiles 0..3
  v8f a10 = {}, a11 = {}, a12 = {}, a13 = {};      // M-subtile 1, N-subtiles 0..3

  union AFrag { v16h v; v8h h[2]; };

  #pragma unroll
  for (int k0 = 0; k0 < KDIM; k0 += 32) {
    AFrag af0, af1;
    af0.h[0] = *(const v8h*)(arow0 + k0 + kbA);       // K = kbA .. kbA+7
    af0.h[1] = *(const v8h*)(arow0 + k0 + kbA + 8);   // K = kbA+8 .. kbA+15
    af1.h[0] = *(const v8h*)(arow1 + k0 + kbA);
    af1.h[1] = *(const v8h*)(arow1 + k0 + kbA + 8);
    v16h b0 = *(const v16h*)(w0 + k0);
    v16h b1 = *(const v16h*)(w1 + k0);
    v16h b2 = *(const v16h*)(w2 + k0);
    v16h b3 = *(const v16h*)(w3 + k0);
    a00 = __builtin_amdgcn_wmma_f32_16x16x32_f16(false, af0.v, false, b0, (short)0, a00, false, false);
    a10 = __builtin_amdgcn_wmma_f32_16x16x32_f16(false, af1.v, false, b0, (short)0, a10, false, false);
    a01 = __builtin_amdgcn_wmma_f32_16x16x32_f16(false, af0.v, false, b1, (short)0, a01, false, false);
    a11 = __builtin_amdgcn_wmma_f32_16x16x32_f16(false, af1.v, false, b1, (short)0, a11, false, false);
    a02 = __builtin_amdgcn_wmma_f32_16x16x32_f16(false, af0.v, false, b2, (short)0, a02, false, false);
    a12 = __builtin_amdgcn_wmma_f32_16x16x32_f16(false, af1.v, false, b2, (short)0, a12, false, false);
    a03 = __builtin_amdgcn_wmma_f32_16x16x32_f16(false, af0.v, false, b3, (short)0, a03, false, false);
    a13 = __builtin_amdgcn_wmma_f32_16x16x32_f16(false, af1.v, false, b3, (short)0, a13, false, false);
  }

  // C layout: VGPR r <-> row mb+r (mb = 0|8 by half-wave), lane <-> column.
  const int mb = (lane < 16) ? 0 : 8;
  const float bj0 = bias[n0 +  0 + mlane];
  const float bj1 = bias[n0 + 16 + mlane];
  const float bj2 = bias[n0 + 32 + mlane];
  const float bj3 = bias[n0 + 48 + mlane];
  float* __restrict__ c0 = C + (size_t)(mt * 32 + mb) * Ncols + n0 + mlane;
  float* __restrict__ c1 = c0 + (size_t)16 * Ncols;
  #pragma unroll
  for (int r = 0; r < 8; ++r) {
    float* crow0 = c0 + (size_t)r * Ncols;
    crow0[ 0] = a00[r] + bj0;
    crow0[16] = a01[r] + bj1;
    crow0[32] = a02[r] + bj2;
    crow0[48] = a03[r] + bj3;
    float* crow1 = c1 + (size_t)r * Ncols;
    crow1[ 0] = a10[r] + bj0;
    crow1[16] = a11[r] + bj1;
    crow1[32] = a12[r] + bj2;
    crow1[48] = a13[r] + bj3;
  }
}

// ---------------------------------------------------------------------------
// Elementwise / scatter helpers
// ---------------------------------------------------------------------------
__global__ void ggnn_zero_f32(float* p, int n) {
  int i = blockIdx.x * blockDim.x + threadIdx.x;
  if (i < n) p[i] = 0.0f;
}

__global__ void ggnn_f32_to_f16(const float* __restrict__ s, _Float16* __restrict__ d, int n) {
  int i = blockIdx.x * blockDim.x + threadIdx.x;
  if (i < n) d[i] = (_Float16)s[i];
}

__global__ void ggnn_init_h(const float* __restrict__ f, float* __restrict__ h,
                            _Float16* __restrict__ h16, int n) {
  int i = blockIdx.x * blockDim.x + threadIdx.x;
  if (i < n) { float v = f[i]; h[i] = v; h16[i] = (_Float16)v; }
}

// one wave per edge: gather a[src] row (256 f32 as float4), atomic-add into m[dst]
__global__ __launch_bounds__(256)
void ggnn_edge_scatter(const float4* __restrict__ a, float* __restrict__ m,
                       const int* __restrict__ src, const int* __restrict__ dst, int E) {
  int e    = (blockIdx.x * blockDim.x + threadIdx.x) >> 5;
  int lane = threadIdx.x & 31;
  if (e >= E) return;
  int s = src[e], d = dst[e];
  const float4* __restrict__ ap = a + (size_t)s * 64;
  float* __restrict__ mp = m + (size_t)d * 256;
  #pragma unroll
  for (int i = 0; i < 2; ++i) {
    int c = lane + i * 32;
    float4 v = ap[c];
    float* q = mp + c * 4;
    atomicAdd(q + 0, v.x);
    atomicAdd(q + 1, v.y);
    atomicAdd(q + 2, v.z);
    atomicAdd(q + 3, v.w);
  }
}

__device__ __forceinline__ float ggnn_sigmoid(float x) { return 1.0f / (1.0f + __expf(-x)); }

// GRUCell: h' = (1-z)*n + z*h  (gi/gh are [N,768] with bias already added)
__global__ void ggnn_gru(const float* __restrict__ gi, const float* __restrict__ gh,
                         float* __restrict__ h, _Float16* __restrict__ h16, int total) {
  int idx = blockIdx.x * blockDim.x + threadIdx.x;
  if (idx >= total) return;
  int i = idx >> 8, j = idx & 255;
  const float* gir = gi + (size_t)i * 768;
  const float* ghr = gh + (size_t)i * 768;
  float r  = ggnn_sigmoid(gir[j]       + ghr[j]);
  float z  = ggnn_sigmoid(gir[256 + j] + ghr[256 + j]);
  float nn = tanhf(gir[512 + j] + r * ghr[512 + j]);
  float hv = h[idx];
  float out = (1.0f - z) * nn + z * hv;
  h[idx]   = out;
  h16[idx] = (_Float16)out;
}

__global__ void ggnn_sigmoid_h(float* __restrict__ h, _Float16* __restrict__ h16, int n) {
  int i = blockIdx.x * blockDim.x + threadIdx.x;
  if (i < n) { float v = ggnn_sigmoid(h[i]); h[i] = v; h16[i] = (_Float16)v; }
}

// ---------------------------------------------------------------------------
// Global attention pooling
// ---------------------------------------------------------------------------
__device__ __forceinline__ unsigned ggnn_fkey(float f) {
  unsigned b = __float_as_uint(f);
  return (b & 0x80000000u) ? ~b : (b | 0x80000000u);   // order-preserving
}
__device__ __forceinline__ float ggnn_fkey_inv(unsigned k) {
  unsigned b = (k & 0x80000000u) ? (k & 0x7FFFFFFFu) : ~k;
  return __uint_as_float(b);
}

// one wave per node: gate = h . Wg + bg ; atomicMax per-graph key
__global__ __launch_bounds__(256)
void ggnn_gate(const float* __restrict__ h, const float* __restrict__ Wg,
               const float* __restrict__ bg, const int* __restrict__ gid,
               float* __restrict__ gate, unsigned* __restrict__ gmaxkey, int Nn) {
  int node = (blockIdx.x * blockDim.x + threadIdx.x) >> 5;
  int lane = threadIdx.x & 31;
  if (node >= Nn) return;
  const float* hr = h + (size_t)node * 256;
  float s = 0.0f;
  #pragma unroll
  for (int j = 0; j < 8; ++j) s += hr[lane + j * 32] * Wg[lane + j * 32];
  #pragma unroll
  for (int off = 16; off > 0; off >>= 1) s += __shfl_down(s, off, 32);
  if (lane == 0) {
    float g = s + bg[0];
    gate[node] = g;
    atomicMax(gmaxkey + gid[node], ggnn_fkey(g));
  }
}

__global__ void ggnn_exp(const float* __restrict__ gate, const unsigned* __restrict__ gmaxkey,
                         const int* __restrict__ gid, float* __restrict__ e,
                         float* __restrict__ denom, int Nn) {
  int i = blockIdx.x * blockDim.x + threadIdx.x;
  if (i >= Nn) return;
  int g = gid[i];
  float ev = __expf(gate[i] - ggnn_fkey_inv(gmaxkey[g]));
  e[i] = ev;
  atomicAdd(denom + g, ev);
}

// one float4 of one node per thread: pooled[g] += h[i]*alpha[i]
__global__ void ggnn_pool(const float4* __restrict__ h, const float* __restrict__ e,
                          const float* __restrict__ denom, const int* __restrict__ gid,
                          float* __restrict__ pooled, int Nn) {
  int idx = blockIdx.x * blockDim.x + threadIdx.x;       // Nn*64 units
  if (idx >= Nn * 64) return;
  int i = idx >> 6, c4 = idx & 63;
  int g = gid[i];
  float alpha = e[i] / denom[g];
  float4 hv = h[idx];
  float* q = pooled + (size_t)g * 256 + c4 * 4;
  atomicAdd(q + 0, hv.x * alpha);
  atomicAdd(q + 1, hv.y * alpha);
  atomicAdd(q + 2, hv.z * alpha);
  atomicAdd(q + 3, hv.w * alpha);
}

// one wave per graph: out = sigmoid(pooled . Wo + bo)
__global__ __launch_bounds__(256)
void ggnn_out(const float* __restrict__ pooled, const float* __restrict__ Wo,
              const float* __restrict__ bo, float* __restrict__ out, int Bn) {
  int b    = (blockIdx.x * blockDim.x + threadIdx.x) >> 5;
  int lane = threadIdx.x & 31;
  if (b >= Bn) return;
  const float* pr = pooled + (size_t)b * 256;
  float s = 0.0f;
  #pragma unroll
  for (int j = 0; j < 8; ++j) s += pr[lane + j * 32] * Wo[lane + j * 32];
  #pragma unroll
  for (int off = 16; off > 0; off >>= 1) s += __shfl_down(s, off, 32);
  if (lane == 0) out[b] = ggnn_sigmoid(s + bo[0]);
}

// ---------------------------------------------------------------------------
// Orchestration
// ---------------------------------------------------------------------------
extern "C" void kernel_launch(void* const* d_in, const int* in_sizes, int n_in,
                              void* d_out, int out_size, void* d_ws, size_t ws_size,
                              hipStream_t stream) {
  (void)n_in; (void)ws_size;
  const float* features = (const float*)d_in[0];
  const int*   src      = (const int*)d_in[1];
  const int*   dst      = (const int*)d_in[2];
  const int*   gid      = (const int*)d_in[3];
  const float* W[2]   = { (const float*)d_in[4],  (const float*)d_in[10] };
  const float* b[2]   = { (const float*)d_in[5],  (const float*)d_in[11] };
  const float* Wih[2] = { (const float*)d_in[6],  (const float*)d_in[12] };
  const float* Whh[2] = { (const float*)d_in[7],  (const float*)d_in[13] };
  const float* bih[2] = { (const float*)d_in[8],  (const float*)d_in[14] };
  const float* bhh[2] = { (const float*)d_in[9],  (const float*)d_in[15] };
  const float* Wg = (const float*)d_in[16];
  const float* bg = (const float*)d_in[17];
  const float* Wo = (const float*)d_in[18];
  const float* bo = (const float*)d_in[19];

  const int Nn = in_sizes[3];     // 40000 nodes
  const int E  = in_sizes[1];     // 400000 edges
  const int Bn = out_size;        // 256 graphs
  const int NF = Nn * 256;        // node-feature elements
  const int NG = Nn * 768;        // gate elements

  // ---- carve workspace (256B-aligned chunks) ----
  uintptr_t base = (uintptr_t)d_ws;
  auto carve = [&](size_t bytes) -> void* {
    void* p = (void*)base;
    base += (bytes + 255) & ~(size_t)255;
    return p;
  };
  float*     h     = (float*)carve((size_t)NF * 4);
  _Float16*  h16   = (_Float16*)carve((size_t)NF * 2);
  float*     a     = (float*)carve((size_t)NF * 4);
  float*     m     = (float*)carve((size_t)NF * 4);
  _Float16*  m16   = (_Float16*)carve((size_t)NF * 2);
  float*     gi    = (float*)carve((size_t)NG * 4);
  float*     gh    = (float*)carve((size_t)NG * 4);
  _Float16*  Wh[2]   = { (_Float16*)carve(256 * 256 * 2), (_Float16*)carve(256 * 256 * 2) };
  _Float16*  Wihh[2] = { (_Float16*)carve(768 * 256 * 2), (_Float16*)carve(768 * 256 * 2) };
  _Float16*  Whhh[2] = { (_Float16*)carve(768 * 256 * 2), (_Float16*)carve(768 * 256 * 2) };
  float*     gate    = (float*)carve((size_t)Nn * 4);
  float*     ev      = (float*)carve((size_t)Nn * 4);
  unsigned*  gmaxkey = (unsigned*)carve((size_t)Bn * 4);
  float*     denom   = (float*)carve((size_t)Bn * 4);
  float*     pooled  = (float*)carve((size_t)Bn * 256 * 4);

  const int T = 256;
  auto blks = [](int n, int t) { return (n + t - 1) / t; };

  // ---- weights -> f16 (recomputed every call; deterministic) ----
  for (int li = 0; li < 2; ++li) {
    ggnn_f32_to_f16<<<blks(256 * 256, T), T, 0, stream>>>(W[li],   Wh[li],   256 * 256);
    ggnn_f32_to_f16<<<blks(768 * 256, T), T, 0, stream>>>(Wih[li], Wihh[li], 768 * 256);
    ggnn_f32_to_f16<<<blks(768 * 256, T), T, 0, stream>>>(Whh[li], Whhh[li], 768 * 256);
  }

  // ---- h = features (IN == OUT, no padding needed) ----
  ggnn_init_h<<<blks(NF, T), T, 0, stream>>>(features, h, h16, NF);

  const int mtiles32 = blks(Nn, 32);           // 1250 (exact: 40000/32)
  dim3 gemmBlk(256);
  dim3 gemm256(blks(mtiles32, 8), 256 / 64);   // a = h@W.T
  dim3 gemm768(blks(mtiles32, 8), 768 / 64);   // gi/gh

  for (int li = 0; li < 2; ++li) {
    for (int step = 0; step < 3; ++step) {
      // a = h @ W.T + b
      ggnn_wmma_gemm<<<gemm256, gemmBlk, 0, stream>>>(h16, Wh[li], b[li], a, Nn, 256);
      // m = segment_sum(a[src], dst)
      ggnn_zero_f32<<<blks(NF, T), T, 0, stream>>>(m, NF);
      ggnn_edge_scatter<<<blks(E * 32, T), T, 0, stream>>>((const float4*)a, m, src, dst, E);
      ggnn_f32_to_f16<<<blks(NF, T), T, 0, stream>>>(m, m16, NF);
      // gi = m @ Wih.T + bih ; gh = h @ Whh.T + bhh
      ggnn_wmma_gemm<<<gemm768, gemmBlk, 0, stream>>>(m16, Wihh[li], bih[li], gi, Nn, 768);
      ggnn_wmma_gemm<<<gemm768, gemmBlk, 0, stream>>>(h16, Whhh[li], bhh[li], gh, Nn, 768);
      // h = GRU(m, h)
      ggnn_gru<<<blks(NF, T), T, 0, stream>>>(gi, gh, h, h16, NF);
    }
    // y = sigmoid(layer(g, x))
    ggnn_sigmoid_h<<<blks(NF, T), T, 0, stream>>>(h, h16, NF);
  }

  // ---- global attention pool + output ----
  ggnn_zero_f32<<<blks(Bn, T), T, 0, stream>>>((float*)gmaxkey, Bn);   // key 0 == -max
  ggnn_zero_f32<<<blks(Bn, T), T, 0, stream>>>(denom, Bn);
  ggnn_zero_f32<<<blks(Bn * 256, T), T, 0, stream>>>(pooled, Bn * 256);
  ggnn_gate<<<blks(Nn * 32, T), T, 0, stream>>>(h, Wg, bg, gid, gate, gmaxkey, Nn);
  ggnn_exp<<<blks(Nn, T), T, 0, stream>>>(gate, gmaxkey, gid, ev, denom, Nn);
  ggnn_pool<<<blks(Nn * 64, T), T, 0, stream>>>((const float4*)h, ev, denom, gid, pooled, Nn);
  ggnn_out<<<blks(Bn * 32, T), T, 0, stream>>>(pooled, Wo, bo, (float*)d_out, Bn);
}
